// NGP_40681930227958
// MI455X (gfx1250) — matrix operand order repaired and checked by reading, not compiled, not run on hardware
//
#include <hip/hip_runtime.h>
#include <hip/hip_bf16.h>
#include <math.h>

// ---------------- CDNA5 WMMA types ----------------
typedef _Float16 v16h __attribute__((ext_vector_type(16)));
typedef float    v8f  __attribute__((ext_vector_type(8)));

#define T_HASH   524288u
#define T_MASKC  524287u
#define HPI2     2654435761u
#define HPI3     805459861u
#define PPB      128          // points per block
#define NTHREADS 256          // 8 waves of 32
#define FEAT_SCALE      64.0f
#define INV_FEAT_SCALE  (1.0f/64.0f)

static __device__ const float NLc[16] = {
    16.f, 22.f, 30.f, 42.f, 58.f, 80.f, 110.f, 152.f,
    209.f, 288.f, 397.f, 547.f, 754.f, 1039.f, 1432.f, 1974.f};

// ceil-selector bitmasks per vertex k (bit k): x:{1,5,6,7} y:{2,4,6,7} z:{3,4,5,7}
#define CXB 0xE2u
#define CYB 0xD4u
#define CZB 0xB8u

// ---------------- fragment loaders (ISA 7.12.2 layouts) ----------------
// A (16xK f16, row-major in LDS): lane L: c=L/16, m=L%16.
// VGPR v holds K = 16*(v/4) + 8c + 2*(v&3) + {0,1} -> merges into 2x ds_load_b128.
__device__ __forceinline__ v16h ldA(const _Float16* A, int stride, int kofs, int lane) {
    union { v16h h; unsigned u[8]; } r;
    const int c = lane >> 4, m = lane & 15;
    const _Float16* p = A + m * stride + kofs + 8 * c;
#pragma unroll
    for (int v = 0; v < 8; ++v) {
        const int k0 = ((v >> 2) << 4) + ((v & 3) << 1);
        r.u[v] = *(const unsigned*)(p + k0);
    }
    return r.h;
}

// B (KxN f16) stored TRANSPOSED in LDS as WT[n][k]. lane L: c=L/16, n=L%16.
// VGPR v holds K = 16*(v/4) + 8c + 2*(v&3) + {0,1} (same order as ldA).
__device__ __forceinline__ v16h ldB(const _Float16* WT, int K, int kofs, int ncol, int lane) {
    union { v16h h; unsigned u[8]; } r;
    const int c = lane >> 4;
    const _Float16* p = WT + ncol * K + kofs + 8 * c;
#pragma unroll
    for (int v = 0; v < 8; ++v) r.u[v] = *(const unsigned*)(p + ((v >> 2) << 4) + ((v & 3) << 1));
    return r.h;
}

__device__ __forceinline__ v8f wmma16(v16h a, v16h b, v8f c) {
    // (neg_a, A, neg_b, B, c_mod, C, reuse_a, reuse_b)
    return __builtin_amdgcn_wmma_f32_16x16x32_f16(false, a, false, b, (short)0, c, false, false);
}

// D epilogue: element r of v8f = row (r + 8c), col n. relu + bias -> f16 LDS [16][64].
__device__ __forceinline__ void st_relu(v8f acc, _Float16* Tw, int colbase,
                                        const float* bias, int lane) {
    const int c = lane >> 4, nn = lane & 15, col = colbase + nn;
    const float b = bias[col];
#pragma unroll
    for (int r = 0; r < 8; ++r)
        Tw[(r + 8 * c) * 64 + col] = (_Float16)fmaxf(acc[r] + b, 0.0f);
}

// One N-tile of a K=64 layer: 2 chained WMMAs + relu/bias epilogue.
// Called with literal nt so every call is straight-line (guaranteed static WMMA sites).
__device__ __forceinline__ void layer64_step(const v16h a0, const v16h a1,
                                             const _Float16* WT, int nt,
                                             _Float16* Tw, const float* bias, int lane) {
    const v16h b0 = ldB(WT, 64, 0, nt * 16 + (lane & 15), lane);
    const v16h b1 = ldB(WT, 64, 32, nt * 16 + (lane & 15), lane);
    v8f acc = {0.f, 0.f, 0.f, 0.f, 0.f, 0.f, 0.f, 0.f};
    acc = wmma16(a0, b0, acc);
    acc = wmma16(a1, b1, acc);
    st_relu(acc, Tw, nt * 16, bias, lane);
}

// One N-tile of the K=32 first layer.
__device__ __forceinline__ void layer32_step(const v16h a, const _Float16* WT, int nt,
                                             _Float16* Tw, const float* bias, int lane) {
    const v16h b = ldB(WT, 32, 0, nt * 16 + (lane & 15), lane);
    v8f acc = {0.f, 0.f, 0.f, 0.f, 0.f, 0.f, 0.f, 0.f};
    acc = wmma16(a, b, acc);
    st_relu(acc, Tw, nt * 16, bias, lane);
}

extern "C" __global__ void __launch_bounds__(NTHREADS)
ngp_fused(const float* __restrict__ X, const float* __restrict__ Dd,
          const float* __restrict__ TBL,
          const float* __restrict__ w1a, const float* __restrict__ b1a,
          const float* __restrict__ w1b, const float* __restrict__ b1b,
          const float* __restrict__ w2a, const float* __restrict__ b2a,
          const float* __restrict__ w2b, const float* __restrict__ b2b,
          const float* __restrict__ w2c, const float* __restrict__ b2c,
          float* __restrict__ outC, float* __restrict__ outS, int npts) {
    // ---- LDS arena (62976 B < 64 KB) ----
    __shared__ __align__(16) unsigned char smem[62976];
    _Float16* W1aT = (_Float16*)(smem + 0);      // [64][32] transposed, pre-scaled 1/64
    _Float16* W1bT = (_Float16*)(smem + 4096);   // [16][64]
    _Float16* W2aT = (_Float16*)(smem + 6144);   // [64][64] (rows 43..63 zero)
    _Float16* W2bT = (_Float16*)(smem + 14336);  // [64][64]
    _Float16* W2cT = (_Float16*)(smem + 22528);  // [16][64] (cols 3..15 zero)
    float*    BIAS = (float*)(smem + 24576);     // b1a[64] b1b[16] b2a[64] b2b[64] b2c[16]
    _Float16* sH   = (_Float16*)(smem + 25472);  // [128][16]
    float*    sH0  = (float*)(smem + 29568);     // [128] f32 h[:,0] for sigma
    unsigned char* sMask = smem + 30080;         // [128]
    _Float16* sT   = (_Float16*)(smem + 30208);  // [8 waves][16][64] scratch
    _Float16* sFeat = (_Float16*)(smem + 46592); // [128][32] (phase 1-2)
    _Float16* sZ    = (_Float16*)(smem + 46592); // [128][64] (phase 3-4, overlapped)

    const int tid  = threadIdx.x;
    const int lane = tid & 31;
    const int wave = tid >> 5;
    const long long blockbase = (long long)blockIdx.x * PPB;

    // ---------- phase 0: weights -> transposed f16 LDS (coalesced global reads) ----------
    // zero-pad passes first (no global traffic, disjoint slots from fill passes)
    for (int s = tid; s < 64 * 64; s += NTHREADS) {       // W2aT rows k>=43 -> 0
        const int kk = s & 63;
        if (kk >= 43) W2aT[s] = (_Float16)0.0f;
    }
    for (int s = tid; s < 16 * 64; s += NTHREADS) {       // W2cT cols n>=3 -> 0
        const int nn = s >> 6;
        if (nn >= 3) W2cT[s] = (_Float16)0.0f;
    }
    // contiguous global index -> transposed LDS slot
    for (int j = tid; j < 32 * 64; j += NTHREADS) {       // w1a [32][64]
        const int kk = j >> 6, nn = j & 63;
        W1aT[nn * 32 + kk] = (_Float16)(w1a[j] * INV_FEAT_SCALE);
    }
    for (int j = tid; j < 64 * 16; j += NTHREADS) {       // w1b [64][16]
        const int kk = j >> 4, nn = j & 15;
        W1bT[nn * 64 + kk] = (_Float16)w1b[j];
    }
    for (int j = tid; j < 43 * 64; j += NTHREADS) {       // w2a [43][64]
        const int kk = j >> 6, nn = j & 63;
        W2aT[nn * 64 + kk] = (_Float16)w2a[j];
    }
    for (int j = tid; j < 64 * 64; j += NTHREADS) {       // w2b [64][64]
        const int kk = j >> 6, nn = j & 63;
        W2bT[nn * 64 + kk] = (_Float16)w2b[j];
    }
    if (tid < 64 * 3) {                                   // w2c [64][3]
        const int kk = tid / 3, nn = tid - 3 * kk;
        W2cT[nn * 64 + kk] = (_Float16)w2c[tid];
    }
    if (tid < 224) {
        float v;
        if (tid < 64)       v = b1a[tid];
        else if (tid < 80)  v = b1b[tid - 64];
        else if (tid < 144) v = b2a[tid - 80];
        else if (tid < 208) v = b2b[tid - 144];
        else                v = (tid - 208 < 3) ? b2c[tid - 208] : 0.0f;
        BIAS[tid] = v;
    }

    // ---------- phase 1: hash-grid features (2 threads split 16 levels of a point) ----------
    {
        const int i = tid & (PPB - 1);
        const int lh = tid >> 7;  // level half: 0 -> levels 0..7, 1 -> 8..15
        long long gp = blockbase + i;
        if (gp >= npts) gp = 0;
        const float x0 = X[gp * 3 + 0] / 3.0f;
        const float x1 = X[gp * 3 + 1] / 3.0f;
        const float x2 = X[gp * 3 + 2] / 3.0f;
        if (lh == 0)
            sMask[i] = (fabsf(x0) < 0.5f && fabsf(x1) < 0.5f && fabsf(x2) < 0.5f) ? 1 : 0;
        const float xu0 = x0 + 0.5f, xu1 = x1 + 0.5f, xu2 = x2 + 0.5f;
        for (int j = 0; j < 8; ++j) {
            const int lvl = lh * 8 + j;
            const float nres = NLc[lvl];
            const float p0 = xu0 * nres, p1 = xu1 * nres, p2 = xu2 * nres;
            const float f0 = floorf(p0), f1 = floorf(p1), f2 = floorf(p2);
            const int fl0 = (int)f0, fl1 = (int)f1, fl2 = (int)f2;
            const int ce0 = (int)ceilf(p0), ce1 = (int)ceilf(p1), ce2 = (int)ceilf(p2);
            const float fr0 = p0 - f0, fr1 = p1 - f1, fr2 = p2 - f2;
            const float* tbase = TBL + (size_t)lvl * (size_t)T_HASH * 2u;
            float a0 = 0.0f, a1 = 0.0f;
#pragma unroll
            for (int k = 0; k < 8; ++k) {
                const unsigned vx = (unsigned)(((CXB >> k) & 1u) ? ce0 : fl0);
                const unsigned vy = (unsigned)(((CYB >> k) & 1u) ? ce1 : fl1);
                const unsigned vz = (unsigned)(((CZB >> k) & 1u) ? ce2 : fl2);
                const unsigned h = (vx ^ (vy * HPI2) ^ (vz * HPI3)) & T_MASKC;
                const float wx = (k & 1) ? fr0 : 1.0f - fr0;
                const float wy = ((k >> 1) & 1) ? fr1 : 1.0f - fr1;
                const float wz = ((k >> 2) & 1) ? fr2 : 1.0f - fr2;
                const float w = wx * wy * wz;
                const float* e = tbase + (size_t)h * 2u;
                a0 += w * e[0];
                a1 += w * e[1];
            }
            sFeat[i * 32 + 2 * lvl + 0] = (_Float16)(a0 * FEAT_SCALE);
            sFeat[i * 32 + 2 * lvl + 1] = (_Float16)(a1 * FEAT_SCALE);
        }
    }
    __syncthreads();

    // ---------- phase 2: layer1a (32->64, relu) + layer1b (64->16) ; one M-tile per wave ----------
    {
        const int base = wave * 16;
        _Float16* Tw = sT + wave * (16 * 64);
        // layer1a: [16x32]*[32x64]  (feature scaling cancels against pre-scaled W1aT)
        const v16h a = ldA(sFeat + base * 32, 32, 0, lane);
        layer32_step(a, W1aT, 0, Tw, BIAS, lane);
        layer32_step(a, W1aT, 1, Tw, BIAS, lane);
        layer32_step(a, W1aT, 2, Tw, BIAS, lane);
        layer32_step(a, W1aT, 3, Tw, BIAS, lane);
        // layer1b: [16x64]*[64x16], no activation
        const v16h a0 = ldA(Tw, 64, 0, lane);
        const v16h a1 = ldA(Tw, 64, 32, lane);
        const v16h b0 = ldB(W1bT, 64, 0, lane & 15, lane);
        const v16h b1 = ldB(W1bT, 64, 32, lane & 15, lane);
        v8f acc = {0.f, 0.f, 0.f, 0.f, 0.f, 0.f, 0.f, 0.f};
        acc = wmma16(a0, b0, acc);
        acc = wmma16(a1, b1, acc);
        const int cg = lane >> 4, nn = lane & 15;
        const float bb = BIAS[64 + nn];
#pragma unroll
        for (int r = 0; r < 8; ++r) {
            const int row = base + r + 8 * cg;
            const float hv = acc[r] + bb;
            sH[row * 16 + nn] = (_Float16)hv;
            if (nn == 0) sH0[row] = hv;  // keep f32 for sigma
        }
    }
    __syncthreads();

    // ---------- phase 3: z = [h(16) | posenc(27) | 0...] ; sigma store ----------
    if (tid < PPB) {
        const int i = tid;
        const long long gp = blockbase + i;
        const bool ok = gp < npts;
        const long long g = ok ? gp : 0;
        const float d0 = Dd[g * 3 + 0], d1 = Dd[g * 3 + 1], d2 = Dd[g * 3 + 2];
        _Float16* z = sZ + i * 64;
#pragma unroll
        for (int k = 0; k < 16; ++k) z[k] = sH[i * 16 + k];
        z[16] = (_Float16)d0; z[17] = (_Float16)d1; z[18] = (_Float16)d2;
        float sc = 1.0f;
#pragma unroll
        for (int e = 0; e < 4; ++e) {
            float s0, c0, s1, c1, s2, c2;
            sincosf(sc * d0, &s0, &c0);
            sincosf(sc * d1, &s1, &c1);
            sincosf(sc * d2, &s2, &c2);
            z[19 + 6 * e + 0] = (_Float16)s0;
            z[19 + 6 * e + 1] = (_Float16)s1;
            z[19 + 6 * e + 2] = (_Float16)s2;
            z[19 + 6 * e + 3] = (_Float16)c0;
            z[19 + 6 * e + 4] = (_Float16)c1;
            z[19 + 6 * e + 5] = (_Float16)c2;
            sc *= 2.0f;
        }
#pragma unroll
        for (int k = 43; k < 64; ++k) z[k] = (_Float16)0.0f;
        if (ok) outS[gp] = sMask[i] ? expf(sH0[i]) : 0.0f;
    }
    __syncthreads();

    // ---------- phase 4: layer2a (43->64 relu), layer2b (64->64 relu), layer2c (64->3 sigmoid) ----------
    {
        const int base = wave * 16;
        _Float16* Tw = sT + wave * (16 * 64);
        // layer2a
        {
            const v16h a0 = ldA(sZ + base * 64, 64, 0, lane);
            const v16h a1 = ldA(sZ + base * 64, 64, 32, lane);
            layer64_step(a0, a1, W2aT, 0, Tw, BIAS + 80, lane);
            layer64_step(a0, a1, W2aT, 1, Tw, BIAS + 80, lane);
            layer64_step(a0, a1, W2aT, 2, Tw, BIAS + 80, lane);
            layer64_step(a0, a1, W2aT, 3, Tw, BIAS + 80, lane);
        }
        // layer2b (in-place: inputs fully fetched to regs before stores)
        {
            const v16h a0 = ldA(Tw, 64, 0, lane);
            const v16h a1 = ldA(Tw, 64, 32, lane);
            layer64_step(a0, a1, W2bT, 0, Tw, BIAS + 144, lane);
            layer64_step(a0, a1, W2bT, 1, Tw, BIAS + 144, lane);
            layer64_step(a0, a1, W2bT, 2, Tw, BIAS + 144, lane);
            layer64_step(a0, a1, W2bT, 3, Tw, BIAS + 144, lane);
        }
        // layer2c -> sigmoid -> masked color
        const v16h e0 = ldA(Tw, 64, 0, lane);
        const v16h e1 = ldA(Tw, 64, 32, lane);
        const v16h b0 = ldB(W2cT, 64, 0, lane & 15, lane);
        const v16h b1 = ldB(W2cT, 64, 32, lane & 15, lane);
        v8f acc = {0.f, 0.f, 0.f, 0.f, 0.f, 0.f, 0.f, 0.f};
        acc = wmma16(e0, b0, acc);
        acc = wmma16(e1, b1, acc);
        const int cg = lane >> 4, nn = lane & 15;
        if (nn < 3) {
            const float bb = BIAS[208 + nn];
#pragma unroll
            for (int r = 0; r < 8; ++r) {
                const int row = base + r + 8 * cg;
                const long long gp = blockbase + row;
                if (gp < npts) {
                    const float v = acc[r] + bb;
                    const float col = 1.0f / (1.0f + expf(-v));
                    outC[gp * 3 + nn] = sMask[row] ? col : 0.0f;
                }
            }
        }
    }
}

extern "C" void kernel_launch(void* const* d_in, const int* in_sizes, int n_in,
                              void* d_out, int out_size, void* d_ws, size_t ws_size,
                              hipStream_t stream) {
    const float* X   = (const float*)d_in[0];
    const float* Dd  = (const float*)d_in[1];
    const float* TBL = (const float*)d_in[2];
    const float* w1a = (const float*)d_in[3];
    const float* b1a = (const float*)d_in[4];
    const float* w1b = (const float*)d_in[5];
    const float* b1b = (const float*)d_in[6];
    const float* w2a = (const float*)d_in[7];
    const float* b2a = (const float*)d_in[8];
    const float* w2b = (const float*)d_in[9];
    const float* b2b = (const float*)d_in[10];
    const float* w2c = (const float*)d_in[11];
    const float* b2c = (const float*)d_in[12];

    const int npts = in_sizes[0] / 3;
    float* outC = (float*)d_out;                  // [N,3] color first
    float* outS = outC + (size_t)npts * 3;        // [N] sigma after

    const int blocks = (npts + PPB - 1) / PPB;
    hipLaunchKernelGGL(ngp_fused, dim3(blocks), dim3(NTHREADS), 0, stream,
                       X, Dd, TBL, w1a, b1a, w1b, b1b, w2a, b2a, w2b, b2b, w2c, b2c,
                       outC, outS, npts);
}
